// SparseMixtureOfExperts_32418413150764
// MI455X (gfx1250) — matrix operand order repaired and checked by reading, not compiled, not run on hardware
//
#include <hip/hip_runtime.h>
#include <hip/hip_bf16.h>

#define B_TOK   8192
#define D_IN    1024
#define HE_DIM  4096
#define O_DIM   1024
#define E_NUM   8
#define HR_DIM  256

#define TM      32      // tokens per expert tile
#define HCHUNK  128     // hidden columns processed per fused chunk
#define XPITCH  1050    // LDS pitch (uint16) for X rows: 525 dwords, 525%64=13 -> spread banks
#define HPITCH  130     // LDS pitch (uint16) for H chunk

typedef float  v8f  __attribute__((ext_vector_type(8)));
typedef __bf16 v16bf __attribute__((ext_vector_type(16)));

union alignas(32) Frag { unsigned u[8]; v16bf v; };

__device__ __forceinline__ float gelu_erf(float x) {
    return 0.5f * x * (1.0f + erff(x * 0.7071067811865476f));
}

__device__ __forceinline__ unsigned short f2bf(float f) {
    union { float f; unsigned u; } a; a.f = f;
    unsigned u = a.u;
    u += 0x7FFFu + ((u >> 16) & 1u);   // round-to-nearest-even
    return (unsigned short)(u >> 16);
}

__device__ __forceinline__ v8f v8f_zero() {
    v8f v;
#pragma unroll
    for (int i = 0; i < 8; ++i) v[i] = 0.0f;
    return v;
}

// ---------------------------------------------------------------------------
// Zero output + per-expert counters
// ---------------------------------------------------------------------------
__global__ void moe_zero(float* __restrict__ out, int n, int* __restrict__ counts) {
    int i = blockIdx.x * blockDim.x + threadIdx.x;
    if (i < n) out[i] = 0.0f;
    if (i < E_NUM) counts[i] = 0;
}

// ---------------------------------------------------------------------------
// Convert x (fp32 row-major) -> packed bf16 pairs (row-major)
// ---------------------------------------------------------------------------
__global__ void moe_cvt_x(const float* __restrict__ x, unsigned* __restrict__ xb, int n2) {
    int i = blockIdx.x * blockDim.x + threadIdx.x;
    if (i >= n2) return;
    unsigned lo = f2bf(x[2 * i]);
    unsigned hi = f2bf(x[2 * i + 1]);
    xb[i] = lo | (hi << 16);
}

// ---------------------------------------------------------------------------
// Convert + swizzle weights into WMMA B-fragment tile order.
// Tile (kt,nt) = 32x16 elements = 256 uint32, laid out [lane][vgpr j]:
//   n = nt*16 + (lane&15);  k = kt*32 + (j>>2)*16 + (lane>>4)*8 + 2*(j&3)  (pair k,k+1 packed)
// Expert-major, then kt-major over nt.
// ---------------------------------------------------------------------------
__global__ void moe_swz_w(const float* __restrict__ w, unsigned* __restrict__ dst,
                          int K, int N, int nexp) {
    long long i   = (long long)blockIdx.x * blockDim.x + threadIdx.x;
    long long per = (long long)K * N / 2;
    if (i >= per * nexp) return;
    int e    = (int)(i / per);
    int rem  = (int)(i % per);
    int tile = rem >> 8;
    int idx  = rem & 255;
    int lane = idx >> 3, j = idx & 7;
    int ntiles = N >> 4;
    int nt = tile % ntiles, kt = tile / ntiles;
    int n = nt * 16 + (lane & 15);
    int k = kt * 32 + ((j >> 2) << 4) + ((lane >> 4) << 3) + ((j & 3) << 1);
    const float* we = w + (long long)e * K * N;
    unsigned lo = f2bf(we[(long long)k * N + n]);
    unsigned hi = f2bf(we[(long long)(k + 1) * N + n]);
    dst[i] = lo | (hi << 16);
}

// ---------------------------------------------------------------------------
// Router: h = gelu(x@rw1+rb1); logits = gelu(h@rw2+rb2); top-2 -> expert lists
// fp32 VALU (tiny: 4.3 GFLOP), 16 tokens per 256-thread block.
// ---------------------------------------------------------------------------
#define RT_TOK 16
__global__ __launch_bounds__(256) void moe_router(
    const float* __restrict__ x,  const float* __restrict__ rw1, const float* __restrict__ rb1,
    const float* __restrict__ rw2, const float* __restrict__ rb2,
    int* __restrict__ counts, int* __restrict__ lists) {
    __shared__ float xr[RT_TOK][D_IN];
    __shared__ float hl[RT_TOK][HR_DIM + 4];
    __shared__ float lg[RT_TOK][E_NUM];
    int tid = threadIdx.x;
    int t0  = blockIdx.x * RT_TOK;

    for (int it = 0; it < RT_TOK * D_IN / 256; ++it) {
        int gi = tid + it * 256;
        int r = gi >> 10, c = gi & 1023;
        xr[r][c] = x[(long long)(t0 + r) * D_IN + c];
    }
    __syncthreads();

    float acc[RT_TOK];
#pragma unroll
    for (int t = 0; t < RT_TOK; ++t) acc[t] = 0.0f;
    for (int d = 0; d < D_IN; ++d) {
        float w = rw1[d * HR_DIM + tid];
#pragma unroll
        for (int t = 0; t < RT_TOK; ++t) acc[t] += xr[t][d] * w;
    }
    float b1 = rb1[tid];
#pragma unroll
    for (int t = 0; t < RT_TOK; ++t) hl[t][tid] = gelu_erf(acc[t] + b1);
    __syncthreads();

    if (tid < RT_TOK * E_NUM) {
        int t = tid >> 3, e = tid & 7;
        float a = 0.0f;
        for (int r = 0; r < HR_DIM; ++r) a += hl[t][r] * rw2[r * E_NUM + e];
        lg[t][e] = gelu_erf(a + rb2[e]);
    }
    __syncthreads();

    if (tid < RT_TOK) {
        int t = tid;
        int b1i = 0; float v1 = lg[t][0];
        for (int e = 1; e < E_NUM; ++e) if (lg[t][e] > v1) { v1 = lg[t][e]; b1i = e; }
        int b2i = -1; float v2 = -3.4e38f;
        for (int e = 0; e < E_NUM; ++e) {
            if (e == b1i) continue;
            if (lg[t][e] > v2) { v2 = lg[t][e]; b2i = e; }
        }
        int tok = t0 + t;
        int p1 = atomicAdd(&counts[b1i], 1); lists[b1i * B_TOK + p1] = tok;
        int p2 = atomicAdd(&counts[b2i], 1); lists[b2i * B_TOK + p2] = tok;
    }
}

// ---------------------------------------------------------------------------
// Fused 2-layer expert MLP with bf16 WMMA, hidden staged in LDS chunks.
// Block = (expert, 32-token tile), 256 threads = 8 waves.
// Each wave owns BOTH 16-token row-tiles so every global B fragment (W1/W2)
// feeds 2 WMMAs -> halves L2 weight traffic; extra A reads come from LDS.
// Stage A: wave w -> hidden col-tile w of the chunk (2 row x 1 col, K=D).
// Stage B: wave w -> out col-tiles w*8..w*8+7 (2 row x 8 col, K=chunk).
// ---------------------------------------------------------------------------
__global__ __launch_bounds__(256, 1) void moe_expert(
    const unsigned* __restrict__ xb,   // [B][D/2] bf16 pairs
    const unsigned* __restrict__ w1f,  // swizzled bf16 ew1 per expert
    const float*    __restrict__ eb1,
    const unsigned* __restrict__ w2f,  // swizzled bf16 ew2 per expert
    const float*    __restrict__ eb2,
    const int* __restrict__ counts, const int* __restrict__ lists,
    float* __restrict__ out) {
    const int TILES = B_TOK / TM;  // 256
    int e    = blockIdx.x / TILES;
    int tile = blockIdx.x % TILES;
    int cnt  = counts[e];
    int base = tile * TM;
    if (base >= cnt) return;

    __shared__ unsigned short Xs[TM * XPITCH];
    __shared__ unsigned short Hs[TM * HPITCH];
    __shared__ int toks[TM];

    int tid  = threadIdx.x;
    int lane = tid & 31;
    int wv   = tid >> 5;

    if (tid < TM) {
        int li = lists[e * B_TOK + base + tid];
        if (li < 0) li = 0; if (li >= B_TOK) li = B_TOK - 1;
        toks[tid] = (base + tid < cnt) ? li : -1;
    }
    __syncthreads();

    {   // stage X rows (bf16) into LDS
        unsigned* Xsu = (unsigned*)Xs;
        for (int it = 0; it < TM * (D_IN / 2) / 256; ++it) {  // 64 iters
            int gi = tid + it * 256;
            int r = gi >> 9, c = gi & 511;
            int tk = toks[r]; if (tk < 0) tk = 0;
            Xsu[r * (XPITCH / 2) + c] = xb[(long long)tk * (D_IN / 2) + c];
        }
    }
    __syncthreads();

    const int half = lane >> 4;
    const int mlo  = lane & 15;
    const long long perW1 = (long long)D_IN * HE_DIM / 2;
    const long long perW2 = (long long)HE_DIM * O_DIM / 2;
    const unsigned* w1e = w1f + (long long)e * perW1;
    const unsigned* w2e = w2f + (long long)e * perW2;
    const int nt1 = HE_DIM / 16;  // 256
    const int nt2 = O_DIM / 16;   // 64

    // acc[r*8 + ci]: OUT fragment for row-tile r, col-tile wv*8+ci
    v8f acc[16];
#pragma unroll
    for (int i = 0; i < 16; ++i) acc[i] = v8f_zero();

    for (int ch = 0; ch < HE_DIM / HCHUNK; ++ch) {  // 32 chunks
        // ---- Stage A: H[32 x 128] = gelu(X @ W1[:, chunk] + b1) ----
        v8f ha0 = v8f_zero(), ha1 = v8f_zero();     // row-tile 0 / 1
        const int ng = ch * 8 + wv;                 // global hidden n-tile
        for (int dkt = 0; dkt < D_IN / 32; ++dkt) {
            Frag a0, a1;
#pragma unroll
            for (int j = 0; j < 8; ++j) {
                int kl = ((j >> 2) << 4) + (half << 3) + ((j & 3) << 1);
                a0.u[j] = *(const unsigned*)(&Xs[(     mlo) * XPITCH + dkt * 32 + kl]);
                a1.u[j] = *(const unsigned*)(&Xs[(16 + mlo) * XPITCH + dkt * 32 + kl]);
            }
            const uint4* bp = (const uint4*)(w1e + (long long)(dkt * nt1 + ng) * 256 + lane * 8);
            Frag b;
            *(uint4*)&b.u[0] = bp[0]; *(uint4*)&b.u[4] = bp[1];
            ha0 = __builtin_amdgcn_wmma_f32_16x16x32_bf16(false, a0.v, false, b.v, (short)0, ha0, false, false);
            ha1 = __builtin_amdgcn_wmma_f32_16x16x32_bf16(false, a1.v, false, b.v, (short)0, ha1, false, false);
        }
        {   // bias + gelu -> bf16 H chunk in LDS (C-layout rows)
            const float* b1v = eb1 + (long long)e * HE_DIM + ch * HCHUNK;
            int cc = wv * 16 + mlo;                 // chunk-local hidden column
            float bb = b1v[cc];
            int row0 = half << 3;
#pragma unroll
            for (int v = 0; v < 8; ++v) {
                Hs[(row0 + v)      * HPITCH + cc] = f2bf(gelu_erf(ha0[v] + bb));
                Hs[(row0 + v + 16) * HPITCH + cc] = f2bf(gelu_erf(ha1[v] + bb));
            }
        }
        __syncthreads();
        // ---- Stage B: OUT[32 x 1024] += H_chunk @ W2[chunk, :] ----
        const int ctbase = wv * 8;
#pragma unroll
        for (int kt2 = 0; kt2 < HCHUNK / 32; ++kt2) {  // 4
            Frag a0, a1;
#pragma unroll
            for (int j = 0; j < 8; ++j) {
                int kl = ((j >> 2) << 4) + (half << 3) + ((j & 3) << 1);
                a0.u[j] = *(const unsigned*)(&Hs[(     mlo) * HPITCH + kt2 * 32 + kl]);
                a1.u[j] = *(const unsigned*)(&Hs[(16 + mlo) * HPITCH + kt2 * 32 + kl]);
            }
            int ktg = ch * 4 + kt2;
#pragma unroll
            for (int ci = 0; ci < 8; ++ci) {
                const uint4* bp = (const uint4*)(w2e + (long long)(ktg * nt2 + ctbase + ci) * 256 + lane * 8);
                Frag b;
                *(uint4*)&b.u[0] = bp[0]; *(uint4*)&b.u[4] = bp[1];
                acc[ci]     = __builtin_amdgcn_wmma_f32_16x16x32_bf16(false, a0.v, false, b.v, (short)0, acc[ci],     false, false);
                acc[8 + ci] = __builtin_amdgcn_wmma_f32_16x16x32_bf16(false, a1.v, false, b.v, (short)0, acc[8 + ci], false, false);
            }
        }
        __syncthreads();
    }

    // ---- Epilogue: bias2 + gelu, atomic-combine (exactly 2 adds per element) ----
    int active = cnt - base; if (active > TM) active = TM;
    {
        const float* b2v = eb2 + (long long)e * O_DIM;
        int row0 = half << 3;
        int ctbase = wv * 8;
        for (int rt = 0; rt < 2; ++rt) {
            for (int ci = 0; ci < 8; ++ci) {
                int col = (ctbase + ci) * 16 + mlo;
                float bb = b2v[col];
#pragma unroll
                for (int v = 0; v < 8; ++v) {
                    int row = rt * 16 + row0 + v;
                    if (row < active) {
                        int tok = toks[row];
                        unsafeAtomicAdd(&out[(long long)tok * O_DIM + col],
                                        gelu_erf(acc[rt * 8 + ci][v] + bb));
                    }
                }
            }
        }
    }
}

// ---------------------------------------------------------------------------
extern "C" void kernel_launch(void* const* d_in, const int* in_sizes, int n_in,
                              void* d_out, int out_size, void* d_ws, size_t ws_size,
                              hipStream_t stream) {
    const float* x   = (const float*)d_in[0];
    const float* rw1 = (const float*)d_in[1];
    const float* rb1 = (const float*)d_in[2];
    const float* rw2 = (const float*)d_in[3];
    const float* rb2 = (const float*)d_in[4];
    const float* ew1 = (const float*)d_in[5];
    const float* eb1 = (const float*)d_in[6];
    const float* ew2 = (const float*)d_in[7];
    const float* eb2 = (const float*)d_in[8];
    float* out = (float*)d_out;

    char* ws = (char*)d_ws;
    int*      counts = (int*)(ws);
    int*      lists  = (int*)(ws + 1024);                                  // 8*8192*4 = 256 KB
    unsigned* xb     = (unsigned*)(ws + (size_t)(1 << 19));                // +16 MB
    unsigned* w1f    = (unsigned*)(ws + (size_t)(1 << 19) + (1u << 24));   // +64 MB
    unsigned* w2f    = (unsigned*)(ws + (size_t)(1 << 19) + (1u << 24) + (1u << 26));

    moe_zero<<<(B_TOK * O_DIM + 255) / 256, 256, 0, stream>>>(out, B_TOK * O_DIM, counts);
    moe_cvt_x<<<(B_TOK * D_IN / 2 + 255) / 256, 256, 0, stream>>>(x, xb, B_TOK * D_IN / 2);
    moe_swz_w<<<(E_NUM * D_IN * HE_DIM / 2 + 255) / 256, 256, 0, stream>>>(ew1, w1f, D_IN, HE_DIM, E_NUM);
    moe_swz_w<<<(E_NUM * HE_DIM * O_DIM / 2 + 255) / 256, 256, 0, stream>>>(ew2, w2f, HE_DIM, O_DIM, E_NUM);
    moe_router<<<B_TOK / RT_TOK, 256, 0, stream>>>(x, rw1, rb1, rw2, rb2, counts, lists);
    moe_expert<<<E_NUM * (B_TOK / TM), 256, 0, stream>>>(xb, w1f, eb1, w2f, eb2, counts, lists, out);
}
